// GroundingDinoHungarianMatcher_11467562680409
// MI455X (gfx1250) — matrix operand order repaired and checked by reading, not compiled, not run on hardware
//
#include <hip/hip_runtime.h>
#include <hip/hip_bf16.h>

// ---------------- problem constants (from reference) ----------------
#define B_  16
#define Q_  900
#define C_  512
#define T_  40
#define L_  91
#define BQ  (B_ * Q_)     // 14400
#define BT  (B_ * T_)     // 640

#define CLASS_COST 2.0f
#define BBOX_COST  5.0f
#define GIOU_COST  2.0f
#define ALPHA_     0.25f
#define EPS_       1e-8f

typedef __attribute__((ext_vector_type(16))) _Float16 v16h;
typedef __attribute__((ext_vector_type(8)))  _Float16 v8h;
typedef __attribute__((ext_vector_type(4)))  _Float16 v4h;
typedef __attribute__((ext_vector_type(8)))  float    v8f;
typedef __attribute__((ext_vector_type(4)))  float    v4f;

union AFrag { v16h v; v8h h[2]; };

// ------------------------------------------------------------------
// Kernel 1: A[i,c] = pos_cost - neg_cost of sigmoid(logits), in f16.
// 4 elements per thread (float4 in, 8-byte f16x4 out).
// ------------------------------------------------------------------
__global__ __launch_bounds__(256) void prep_focal(const float* __restrict__ logits,
                                                  _Float16* __restrict__ A) {
  const size_t idx = ((size_t)blockIdx.x * blockDim.x + threadIdx.x) * 4;
  v4f x = *(const v4f*)(logits + idx);
  v4h o;
#pragma unroll
  for (int k = 0; k < 4; ++k) {
    float p   = 1.0f / (1.0f + __expf(-x[k]));
    float omp = 1.0f - p;
    float pos = ALPHA_ * omp * omp * (-__logf(p + EPS_));
    float neg = (1.0f - ALPHA_) * p * p * (-__logf(omp + EPS_));
    o[k] = (_Float16)(pos - neg);
  }
  *(v4h*)(A + idx) = o;
}

// ------------------------------------------------------------------
// Kernel 2: gather label_maps rows by class_labels, row-normalize,
// emit LM[row, c] in f16 (row-major [BT, C]). One wave per row.
// ------------------------------------------------------------------
__global__ __launch_bounds__(256) void prep_lm(const float* __restrict__ label_maps,
                                               const int* __restrict__ class_labels,
                                               _Float16* __restrict__ LM) {
  const int wave = (blockIdx.x * (blockDim.x >> 5)) + (threadIdx.x >> 5); // 0..BT-1
  const int lane = threadIdx.x & 31;
  const int b    = wave / T_;
  const int cls  = class_labels[wave];
  const float* src = label_maps + ((size_t)b * L_ + cls) * C_;

  float vals[C_ / 32];
  float s = 0.0f;
#pragma unroll
  for (int k = 0; k < C_ / 32; ++k) {
    vals[k] = src[lane + k * 32];
    s += vals[k];
  }
#pragma unroll
  for (int off = 16; off >= 1; off >>= 1) s += __shfl_xor(s, off, 32);
  const float inv = 1.0f / s;

  _Float16* dst = LM + (size_t)wave * C_;
#pragma unroll
  for (int k = 0; k < C_ / 32; ++k)
    dst[lane + k * 32] = (_Float16)(vals[k] * inv);
}

// ------------------------------------------------------------------
// Kernel 3: WMMA GEMM (class cost) fused with bbox-L1 + GIoU epilogue.
// Each wave: one 16-row M-tile x 4 consecutive 16-col N-tiles.
//   waves = 56 * (40/4) = 560 ; 8 waves/block -> 70 blocks.
// ------------------------------------------------------------------
__global__ __launch_bounds__(256) void matcher_main(const _Float16* __restrict__ A,
                                                    const _Float16* __restrict__ LM,
                                                    const float* __restrict__ pred_boxes,
                                                    const float* __restrict__ boxes,
                                                    float* __restrict__ out) {
  const int lane  = threadIdx.x & 31;
  const int wave  = (blockIdx.x * (blockDim.x >> 5)) + (threadIdx.x >> 5);
  const int mTile = wave / 10;        // 0..55
  const int nT0   = (wave % 10) * 4;  // first of 4 N-tiles

  const int  lmod = lane & 15;
  const bool hi   = (lane & 16) != 0;

  // A fragment addressing (16-bit A 16x32 layout):
  //  lanes 0-15: M=lane, K in {0..7} U {16..23};  lanes 16-31: M=lane-16, K in {8..15} U {24..31}
  const _Float16* aRow = A + (size_t)(mTile * 16 + lmod) * C_ + (hi ? 8 : 0);
  // B fragment addressing (16-bit B 32x16 layout):
  //  lanes 0-15: N=lane, K in {0..7} U {16..23};  lanes 16-31: N=lane-16, K in {8..15} U {24..31}
  const int kloB = hi ? 8 : 0;

  v8f acc[4];
#pragma unroll
  for (int n = 0; n < 4; ++n) acc[n] = (v8f){};

  for (int k0 = 0; k0 < C_; k0 += 32) {
    AFrag af;
    af.h[0] = *(const v8h*)(aRow + k0);
    af.h[1] = *(const v8h*)(aRow + k0 + 16);
#pragma unroll
    for (int n = 0; n < 4; ++n) {
      const _Float16* bp = LM + (size_t)((nT0 + n) * 16 + lmod) * C_ + kloB + k0;
      AFrag bf;
      bf.h[0] = *(const v8h*)bp;
      bf.h[1] = *(const v8h*)(bp + 16);
      acc[n] = __builtin_amdgcn_wmma_f32_16x16x32_f16(
          /*neg_a=*/false, af.v, /*neg_b=*/false, bf.v,
          /*c_mod=*/(short)0, acc[n], /*reuse_a=*/false, /*reuse_b=*/false);
    }
  }

  // ---- epilogue: C/D layout -> VGPR r: M = r + (hi?8:0), N = lmod ----
  const int rOff = hi ? 8 : 0;

  // per-lane pred boxes for the 8 rows this lane owns
  float pcx[8], pcy[8], pw[8], ph[8], p0x[8], p0y[8], p1x[8], p1y[8], areaP[8];
#pragma unroll
  for (int r = 0; r < 8; ++r) {
    const int i = mTile * 16 + r + rOff;
    v4f pb = *(const v4f*)(pred_boxes + (size_t)i * 4);
    pcx[r] = pb[0]; pcy[r] = pb[1]; pw[r] = pb[2]; ph[r] = pb[3];
    p0x[r] = pb[0] - 0.5f * pb[2]; p0y[r] = pb[1] - 0.5f * pb[3];
    p1x[r] = pb[0] + 0.5f * pb[2]; p1y[r] = pb[1] + 0.5f * pb[3];
    areaP[r] = pb[2] * pb[3];
  }

#pragma unroll
  for (int n = 0; n < 4; ++n) {
    const int j = (nT0 + n) * 16 + lmod;
    v4f tb = *(const v4f*)(boxes + (size_t)j * 4);
    const float t0x = tb[0] - 0.5f * tb[2], t0y = tb[1] - 0.5f * tb[3];
    const float t1x = tb[0] + 0.5f * tb[2], t1y = tb[1] + 0.5f * tb[3];
    const float areaT = tb[2] * tb[3];
#pragma unroll
    for (int r = 0; r < 8; ++r) {
      const int i = mTile * 16 + r + rOff;
      const float l1 = fabsf(pcx[r] - tb[0]) + fabsf(pcy[r] - tb[1]) +
                       fabsf(pw[r]  - tb[2]) + fabsf(ph[r]  - tb[3]);
      const float iw = fmaxf(fminf(p1x[r], t1x) - fmaxf(p0x[r], t0x), 0.0f);
      const float ih = fmaxf(fminf(p1y[r], t1y) - fmaxf(p0y[r], t0y), 0.0f);
      const float inter = iw * ih;
      const float uni   = areaP[r] + areaT - inter;
      const float iou   = inter / uni;
      const float ex = fmaxf(fmaxf(p1x[r], t1x) - fminf(p0x[r], t0x), 0.0f);
      const float ey = fmaxf(fmaxf(p1y[r], t1y) - fminf(p0y[r], t0y), 0.0f);
      const float areaE = ex * ey;
      const float giou  = iou - (areaE - uni) / areaE;
      const float cost  = BBOX_COST * l1 + CLASS_COST * acc[n][r] - GIOU_COST * giou;
      out[(size_t)i * BT + j] = cost;
    }
  }
}

// ------------------------------------------------------------------
extern "C" void kernel_launch(void* const* d_in, const int* in_sizes, int n_in,
                              void* d_out, int out_size, void* d_ws, size_t ws_size,
                              hipStream_t stream) {
  (void)in_sizes; (void)n_in; (void)out_size; (void)ws_size;
  const float* logits      = (const float*)d_in[0];
  const float* pred_boxes  = (const float*)d_in[1];
  const float* label_maps  = (const float*)d_in[2];
  const float* boxes       = (const float*)d_in[3];
  const int*   class_labels= (const int*)d_in[4];
  float* out = (float*)d_out;

  _Float16* A  = (_Float16*)d_ws;                                  // [BQ, C] f16 = 14.7 MB
  _Float16* LM = (_Float16*)((char*)d_ws + (size_t)BQ * C_ * 2);   // [BT, C] f16 = 0.64 MB

  // Kernel 1: focal transform -> A (14400*512 / 4 per thread / 256 = 7200 blocks)
  prep_focal<<<(BQ * C_) / 4 / 256, 256, 0, stream>>>(logits, A);
  // Kernel 2: gather+normalize label maps -> LM (640 waves, 8 waves/block)
  prep_lm<<<BT / 8, 256, 0, stream>>>(label_maps, class_labels, LM);
  // Kernel 3: fused WMMA GEMM + bbox/GIoU epilogue (560 waves, 8/block)
  matcher_main<<<(Q_ / 16) * (BT / 64) / 8, 256, 0, stream>>>(A, LM, pred_boxes, boxes, out);
}
